// ViT_LowRank_9371618640660
// MI455X (gfx1250) — matrix operand order repaired and checked by reading, not compiled, not run on hardware
//
#include <hip/hip_runtime.h>
#include <hip/hip_bf16.h>
#include <math.h>

// ---------------- model dims ----------------
#define BATCH   256
#define SEQL    257           // 16*16 patches + cls
#define HID     128
#define NH      2
#define DH      64
#define FF      512
#define OUTD    1000
#define INPUTD  588           // 3*14*14
#define KPAD    608           // mapW K padded to multiple of 32
#define SPAD    288           // padded seq for K-dim of P@V (multiple of 32)
#define KT      17            // key tiles: 17*16 = 272 >= 257
#define NTOK    65792L        // BATCH*SEQL

typedef __attribute__((ext_vector_type(16))) _Float16 v16h;
typedef __attribute__((ext_vector_type(8)))  float    v8f;

static __device__ __forceinline__ v8f vzero() { v8f z = {}; return z; }

static __device__ __forceinline__ v8f wmma16(v16h a, v16h b, v8f c) {
  // v_wmma_f32_16x16x32_f16  D = A*B + C  (fp32 accumulate)
  return __builtin_amdgcn_wmma_f32_16x16x32_f16(false, a, false, b, (short)0, c,
                                                false, false);
}

// ---- A-operand loader: 16x32 f16 tile, row-major source, stride lda (elems).
// Lane layout (ISA 7.12.2): lanes 0-15 row m=lane, K = [0,8)+[16,24);
// lanes 16-31 row m=lane-16, K = [8,16)+[24,32).
static __device__ __forceinline__ v16h load_a_tile(const _Float16* base, int lda) {
  int l = threadIdx.x & 31;
  int row  = l & 15;
  int koff = (l & 16) ? 8 : 0;
  const _Float16* p = base + (long)row * lda + koff;
  union { v16h v; uint4 u[2]; } r;
  r.u[0] = *reinterpret_cast<const uint4*>(p);       // K koff..koff+7
  r.u[1] = *reinterpret_cast<const uint4*>(p + 16);  // K koff+16..koff+23
  return r.v;
}

static __device__ __forceinline__ v16h load_a_tile_f32(const float* base, long lda) {
  int l = threadIdx.x & 31;
  int row  = l & 15;
  int koff = (l & 16) ? 8 : 0;
  const float* p = base + (long)row * lda + koff;
  v16h a;
#pragma unroll
  for (int i = 0; i < 8; ++i) { a[i] = (_Float16)p[i]; a[8 + i] = (_Float16)p[16 + i]; }
  return a;
}

// ---- B-operand loader: 32x16 (KxN) tile, source stored N-major (row n = output
// feature, contiguous over K), stride ldb. Lane n<16 holds K [0,8)+[16,24) of column n,
// lanes 16-31 hold K [8,16)+[24,32).
static __device__ __forceinline__ v16h load_b_tile(const _Float16* base, int ldb) {
  int l = threadIdx.x & 31;
  int n    = l & 15;
  int koff = (l & 16) ? 8 : 0;
  const _Float16* p = base + (long)n * ldb + koff;
  union { v16h v; uint4 u[2]; } r;
  r.u[0] = *reinterpret_cast<const uint4*>(p);
  r.u[1] = *reinterpret_cast<const uint4*>(p + 16);
  return r.v;
}

static __device__ __forceinline__ v16h load_b_tile_cl(const _Float16* base, int ldb,
                                                      int n0, int nmax) {
  int l = threadIdx.x & 31;
  int n = n0 + (l & 15); if (n > nmax) n = nmax;
  int koff = (l & 16) ? 8 : 0;
  const _Float16* p = base + (long)n * ldb + koff;
  union { v16h v; uint4 u[2]; } r;
  r.u[0] = *reinterpret_cast<const uint4*>(p);
  r.u[1] = *reinterpret_cast<const uint4*>(p + 16);
  return r.v;
}

// sinusoidal positional embedding (POS is not an input; built once into ws)
static __device__ __forceinline__ float pos_emb(int s, int j) {
  float jj = (float)(j & ~1);
  float freq = __powf(10000.0f, -jj * (1.0f / 128.0f));
  float ang = (float)s * freq;
  return (j & 1) ? __cosf(ang) : __sinf(ang);
}

// ---------------- prep kernels ----------------

__global__ void k_zero(uint4* p, long n4) {
  long i = (long)blockIdx.x * blockDim.x + threadIdx.x;
  long stride = (long)gridDim.x * blockDim.x;
  uint4 z = make_uint4(0u, 0u, 0u, 0u);
  for (; i < n4; i += stride) p[i] = z;
}

__global__ void k_pos(float* pos) {               // grid = SEQL, block = 128
  int s = blockIdx.x, j = threadIdx.x;
  pos[(long)s * HID + j] = pos_emb(s, j);
}

// plain f32 -> f16 weight conversion (grid-stride)
__global__ void k_cvt(const float* __restrict__ src, _Float16* __restrict__ dst, long n) {
  long i = (long)blockIdx.x * blockDim.x + threadIdx.x;
  long stride = (long)gridDim.x * blockDim.x;
  for (; i < n; i += stride) dst[i] = (_Float16)src[i];
}

// mapW (128 x 588) -> f16 padded to (128 x 608), tail zeros
__global__ void k_cvt_map(const float* __restrict__ src, _Float16* __restrict__ dst) {
  long i = (long)blockIdx.x * blockDim.x + threadIdx.x;
  long stride = (long)gridDim.x * blockDim.x;
  for (; i < (long)HID * KPAD; i += stride) {
    int row = (int)(i / KPAD), kk = (int)(i - (long)row * KPAD);
    dst[i] = (kk < INPUTD) ? (_Float16)src[(long)row * INPUTD + kk] : (_Float16)0.f;
  }
}

// ---------------- model kernels ----------------

// patch embed: tokens = patches @ mapW^T + mapb + POS  -> out[b][1+pid][:]
__global__ __launch_bounds__(256) void k_patch(const float* __restrict__ images,
                                               const _Float16* __restrict__ mapW16,
                                               const float* __restrict__ mapb,
                                               const float* __restrict__ pos,
                                               float* __restrict__ out) {
  __shared__ __align__(16) _Float16 As[128][32];
  const int tid = threadIdx.x;
  const int wv  = tid >> 5;
  const long t0 = (long)blockIdx.x * 128;   // 512 blocks * 128 patch tokens = 65536

  // per-thread gather state: thread owns row (tid>>1), K lane-half (tid&1)*16
  const int arow  = tid >> 1;
  const int kbase = (tid & 1) * 16;
  {
    long token = t0 + arow;
    // nothing: computed below
  }
  long token = t0 + arow;
  int tb = (int)(token >> 8), pid = (int)(token & 255);
  int py = pid >> 4, px = pid & 15;
  const float* img = images + ((long)tb * 3) * 50176 + (long)(py * 14) * 224 + (px * 14);

  v8f acc[8];
#pragma unroll
  for (int i = 0; i < 8; ++i) acc[i] = vzero();

  for (int k0 = 0; k0 < KPAD; k0 += 32) {   // 19 K-steps, zero padded
    __syncthreads();
    {
      int kcur = k0 + kbase;
      int c  = kcur / 196;
      int rem = kcur - c * 196;
      int ir = rem / 14;
      int jc = rem - ir * 14;
#pragma unroll
      for (int it = 0; it < 16; ++it) {
        float av = (kcur < INPUTD) ? img[(long)c * 50176 + ir * 224 + jc] : 0.f;
        As[arow][kbase + it] = (_Float16)av;
        ++kcur;
        if (++jc == 14) { jc = 0; if (++ir == 14) { ir = 0; ++c; } }
      }
    }
    __syncthreads();
    v16h a = load_a_tile(&As[wv * 16][0], 32);
#pragma unroll
    for (int nt = 0; nt < 8; ++nt) {
      v16h b = load_b_tile(mapW16 + (long)(nt * 16) * KPAD + k0, KPAD);
      acc[nt] = wmma16(a, b, acc[nt]);
    }
  }

  int l  = tid & 31;
  int nl = l & 15;
  int mb = (l & 16) ? 8 : 0;
#pragma unroll
  for (int nt = 0; nt < 8; ++nt) {
    int n = nt * 16 + nl;
    float bias = mapb[n];
#pragma unroll
    for (int r = 0; r < 8; ++r) {
      long tok = t0 + wv * 16 + mb + r;
      int b = (int)(tok >> 8), pp = (int)(tok & 255);
      int s = 1 + pp;
      out[((long)b * SEQL + s) * HID + n] = acc[nt][r] + bias + pos[(long)s * HID + n];
    }
  }
}

__global__ void k_cls(const float* __restrict__ cls, const float* __restrict__ pos,
                      float* __restrict__ out) {
  int b = blockIdx.x, j = threadIdx.x;   // block = 128 threads
  out[(long)b * SEQL * HID + j] = cls[j] + pos[j];
}

// layernorm over HID=128, one token per wave, f16 output
__global__ __launch_bounds__(256) void k_ln(const float* __restrict__ in,
                                            const float* __restrict__ g,
                                            const float* __restrict__ bt,
                                            _Float16* __restrict__ dst) {
  int wv = threadIdx.x >> 5, l = threadIdx.x & 31;
  long token = (long)blockIdx.x * 8 + wv;
  if (token >= NTOK) return;
  const float* x = in + token * HID;
  float v[4], s = 0.f;
#pragma unroll
  for (int i = 0; i < 4; ++i) { v[i] = x[i * 32 + l]; s += v[i]; }
#pragma unroll
  for (int off = 16; off; off >>= 1) s += __shfl_xor(s, off, 32);
  float mu = s * (1.0f / 128.0f);
  float var = 0.f;
#pragma unroll
  for (int i = 0; i < 4; ++i) { float d = v[i] - mu; var += d * d; }
#pragma unroll
  for (int off = 16; off; off >>= 1) var += __shfl_xor(var, off, 32);
  float rs = rsqrtf(var * (1.0f / 128.0f) + 1e-5f);
#pragma unroll
  for (int i = 0; i < 4; ++i) {
    int f = i * 32 + l;
    dst[token * HID + f] = (_Float16)((v[i] - mu) * rs * g[f] + bt[f]);
  }
}

// per-head Q,K,V projections (f16 weights); V written transposed
__global__ __launch_bounds__(256) void k_qkv(const _Float16* __restrict__ xln,
                                             const _Float16* __restrict__ qW16,
                                             const _Float16* __restrict__ kW16,
                                             const _Float16* __restrict__ vW16,
                                             const float* __restrict__ qb,
                                             const float* __restrict__ kb,
                                             const float* __restrict__ vb,
                                             _Float16* __restrict__ qbuf,
                                             _Float16* __restrict__ kbuf,
                                             _Float16* __restrict__ vT) {
  int wv = threadIdx.x >> 5, l = threadIdx.x & 31;
  long t0 = ((long)blockIdx.x * 8 + wv) * 16;      // 514 blocks * 8 waves * 16 = 65792
  int nl = l & 15, mb = (l & 16) ? 8 : 0;

  // hoist token -> (batch, seq) decomposition for the 8 C rows
  long bhbase[8]; int sv[8];
#pragma unroll
  for (int r = 0; r < 8; ++r) {
    long t = t0 + mb + r;
    long bb = t / SEQL;
    sv[r] = (int)(t - bb * SEQL);
    bhbase[r] = bb * NH;
  }

#pragma unroll
  for (int h = 0; h < NH; ++h) {
    v16h a0 = load_a_tile(xln + t0 * HID + h * DH + 0, HID);
    v16h a1 = load_a_tile(xln + t0 * HID + h * DH + 32, HID);
    for (int m = 0; m < 3; ++m) {
      const _Float16* W = (m == 0 ? qW16 : m == 1 ? kW16 : vW16) + (long)h * DH * DH;
      const float*    B = (m == 0 ? qb   : m == 1 ? kb   : vb)   + h * DH;
#pragma unroll
      for (int nt = 0; nt < 4; ++nt) {
        v16h b0 = load_b_tile(W + (long)(nt * 16) * DH + 0, DH);
        v16h b1 = load_b_tile(W + (long)(nt * 16) * DH + 32, DH);
        v8f c = vzero();
        c = wmma16(a0, b0, c);
        c = wmma16(a1, b1, c);
        int e = nt * 16 + nl;
        float bias = B[e];
#pragma unroll
        for (int r = 0; r < 8; ++r) {
          long bh = bhbase[r] + h;
          _Float16 val = (_Float16)(c[r] + bias);
          if (m == 0)      qbuf[(bh * SPAD + sv[r]) * DH + e] = val;
          else if (m == 1) kbuf[(bh * SPAD + sv[r]) * DH + e] = val;
          else             vT[(bh * DH + e) * SPAD + sv[r]] = val;
        }
      }
    }
  }
}

// attention for one (b,h); each wave owns one 16-row q-tile; full scores in LDS
__global__ __launch_bounds__(64) void k_attn(const _Float16* __restrict__ qbuf,
                                             const _Float16* __restrict__ kbuf,
                                             const _Float16* __restrict__ vT,
                                             float* __restrict__ out) {
  __shared__ __align__(16) float    Ssc[2][16][272];
  __shared__ __align__(16) _Float16 Pb [2][16][SPAD];
  int wv = threadIdx.x >> 5, l = threadIdx.x & 31;
  int bh = blockIdx.x;
  int b = bh >> 1, h = bh & 1;
  int qt = blockIdx.y * 2 + wv;
  if (qt >= KT) return;                      // wave-uniform
  int s0 = qt * 16;
  const _Float16* qp = qbuf + ((long)bh * SPAD + s0) * DH;
  v16h qa0 = load_a_tile(qp, DH);
  v16h qa1 = load_a_tile(qp + 32, DH);
  int nl = l & 15, mb = (l & 16) ? 8 : 0;

  // S = (Q K^T) * 1/sqrt(64), masked past 257
  for (int kt = 0; kt < KT; ++kt) {
    const _Float16* kp = kbuf + ((long)bh * SPAD + kt * 16) * DH;
    v16h kb0 = load_b_tile(kp, DH);
    v16h kb1 = load_b_tile(kp + 32, DH);
    v8f c = vzero();
    c = wmma16(qa0, kb0, c);
    c = wmma16(qa1, kb1, c);
    int n = kt * 16 + nl;
    bool valid = (n < SEQL);
#pragma unroll
    for (int r = 0; r < 8; ++r)
      Ssc[wv][mb + r][n] = valid ? c[r] * 0.125f : -1e30f;
  }

  // rowwise softmax (lane pair (r, r+16) shares a row; halves of 136 cols each)
  {
    int row = l & 15, hs = l >> 4;
    int c0 = hs * 136;
    float mx = -1e30f;
    for (int j = 0; j < 136; ++j) mx = fmaxf(mx, Ssc[wv][row][c0 + j]);
    mx = fmaxf(mx, __shfl_xor(mx, 16, 32));
    float sum = 0.f;
    for (int j = 0; j < 136; ++j) sum += __expf(Ssc[wv][row][c0 + j] - mx);
    sum += __shfl_xor(sum, 16, 32);
    float inv = 1.0f / sum;
    for (int j = 0; j < 136; ++j)
      Pb[wv][row][c0 + j] = (_Float16)(__expf(Ssc[wv][row][c0 + j] - mx) * inv);
    if (hs) for (int j = 272; j < SPAD; ++j) Pb[wv][row][j] = (_Float16)0.f;
  }

  // O = P @ V  (vT rows contiguous over keys), residual add into out
#pragma unroll
  for (int nt = 0; nt < 4; ++nt) {
    v8f c = vzero();
    for (int ks = 0; ks < SPAD / 32; ++ks) {
      v16h pa = load_a_tile(&Pb[wv][0][ks * 32], SPAD);
      v16h bv = load_b_tile(vT + ((long)bh * DH + nt * 16) * SPAD + ks * 32, SPAD);
      c = wmma16(pa, bv, c);
    }
    int e = nt * 16 + nl;
#pragma unroll
    for (int r = 0; r < 8; ++r) {
      int s = s0 + mb + r;
      if (s < SEQL) {
        long idx = ((long)b * SEQL + s) * HID + h * DH + e;
        out[idx] += c[r];
      }
    }
  }
}

// fc1 + exact GELU -> mlph (f16); wave owns a 16-token tile, N=512
__global__ __launch_bounds__(256) void k_mlp1(const _Float16* __restrict__ xln,
                                              const _Float16* __restrict__ w1,
                                              const float* __restrict__ b1,
                                              _Float16* __restrict__ mlph) {
  int wv = threadIdx.x >> 5, l = threadIdx.x & 31;
  long t0 = ((long)blockIdx.x * 8 + wv) * 16;
  __builtin_prefetch(w1, 0, 2);
  v16h a[4];
#pragma unroll
  for (int ks = 0; ks < 4; ++ks) a[ks] = load_a_tile(xln + t0 * HID + ks * 32, HID);
  int nl = l & 15, mb = (l & 16) ? 8 : 0;
  for (int np = 0; np < 4; ++np) {
    v8f acc[8];
#pragma unroll
    for (int j = 0; j < 8; ++j) acc[j] = vzero();
#pragma unroll
    for (int ks = 0; ks < 4; ++ks) {
#pragma unroll
      for (int j = 0; j < 8; ++j) {
        int nt = np * 8 + j;
        v16h b = load_b_tile(w1 + (long)(nt * 16) * HID + ks * 32, HID);
        acc[j] = wmma16(a[ks], b, acc[j]);
      }
    }
#pragma unroll
    for (int j = 0; j < 8; ++j) {
      int n = (np * 8 + j) * 16 + nl;
      float bias = b1[n];
#pragma unroll
      for (int r = 0; r < 8; ++r) {
        float x = acc[j][r] + bias;
        float gel = 0.5f * x * (1.0f + erff(x * 0.70710678118654752f));
        mlph[(t0 + mb + r) * FF + n] = (_Float16)gel;
      }
    }
  }
}

// fc2 + residual add into out
__global__ __launch_bounds__(256) void k_mlp2(const _Float16* __restrict__ mlph,
                                              const _Float16* __restrict__ w2,
                                              const float* __restrict__ b2,
                                              float* __restrict__ out) {
  int wv = threadIdx.x >> 5, l = threadIdx.x & 31;
  long t0 = ((long)blockIdx.x * 8 + wv) * 16;
  __builtin_prefetch(w2, 0, 2);
  int nl = l & 15, mb = (l & 16) ? 8 : 0;
  v8f acc[8];
#pragma unroll
  for (int j = 0; j < 8; ++j) acc[j] = vzero();
  for (int ks = 0; ks < 16; ++ks) {
    v16h a = load_a_tile(mlph + t0 * FF + ks * 32, FF);
#pragma unroll
    for (int j = 0; j < 8; ++j) {
      v16h b = load_b_tile(w2 + (long)(j * 16) * FF + ks * 32, FF);
      acc[j] = wmma16(a, b, acc[j]);
    }
  }
#pragma unroll
  for (int j = 0; j < 8; ++j) {
    int n = j * 16 + nl;
    float bias = b2[n];
#pragma unroll
    for (int r = 0; r < 8; ++r) out[(t0 + mb + r) * HID + n] += acc[j][r] + bias;
  }
}

// head: logits = cls_out @ headW^T + headb   (M=256 batch rows, N padded to 1008)
__global__ __launch_bounds__(256) void k_head(const float* __restrict__ out,
                                              const _Float16* __restrict__ headW16,
                                              const float* __restrict__ headb,
                                              float* __restrict__ logits) {
  int wv = threadIdx.x >> 5, l = threadIdx.x & 31;
  int m0 = blockIdx.x * 16;                 // 16 blocks * 16 batch rows
  v16h a[4];
#pragma unroll
  for (int ks = 0; ks < 4; ++ks)
    a[ks] = load_a_tile_f32(out + (long)m0 * SEQL * HID + ks * 32, (long)SEQL * HID);
  int nl = l & 15, mb = (l & 16) ? 8 : 0;
  for (int i = 0; i < 8; ++i) {
    int nt = wv + i * 8;
    if (nt >= 63) break;                    // 63 tiles cover 1008 >= 1000
    v8f c = vzero();
#pragma unroll
    for (int ks = 0; ks < 4; ++ks) {
      v16h b = load_b_tile_cl(headW16 + ks * 32, HID, nt * 16, OUTD - 1);
      c = wmma16(a[ks], b, c);
    }
    int n = nt * 16 + nl;
    if (n < OUTD) {
      float bias = headb[n];
#pragma unroll
      for (int r = 0; r < 8; ++r)
        logits[(long)(m0 + mb + r) * OUTD + n] = c[r] + bias;
    }
  }
}

__global__ __launch_bounds__(256) void k_softmax(const float* __restrict__ logits,
                                                 float* __restrict__ outp) {
  __shared__ float red[256];
  int b = blockIdx.x, tid = threadIdx.x;
  const float* row = logits + (long)b * OUTD;
  float mx = -1e30f;
  for (int j = tid; j < OUTD; j += 256) mx = fmaxf(mx, row[j]);
  red[tid] = mx; __syncthreads();
  for (int off = 128; off; off >>= 1) {
    if (tid < off) red[tid] = fmaxf(red[tid], red[tid + off]);
    __syncthreads();
  }
  mx = red[0]; __syncthreads();
  float s = 0.f;
  for (int j = tid; j < OUTD; j += 256) s += __expf(row[j] - mx);
  red[tid] = s; __syncthreads();
  for (int off = 128; off; off >>= 1) {
    if (tid < off) red[tid] += red[tid + off];
    __syncthreads();
  }
  float inv = 1.0f / red[0];
  for (int j = tid; j < OUTD; j += 256)
    outp[(long)b * OUTD + j] = __expf(row[j] - mx) * inv;
}

// ---------------- workspace layout ----------------
static constexpr size_t SZ_OUT  = (size_t)NTOK * HID * 4;          // fp32 residual
static constexpr size_t SZ_XLN  = (size_t)NTOK * HID * 2;          // f16 LN output
static constexpr size_t SZ_QKV  = (size_t)BATCH * NH * SPAD * DH * 2;
static constexpr size_t SZ_MLPH = (size_t)NTOK * FF * 2;
static constexpr size_t SZ_LOG  = (size_t)BATCH * OUTD * 4;
static constexpr size_t SZ_POS  = (size_t)SEQL * HID * 4;
static constexpr size_t SZ_MAP16 = (size_t)HID * KPAD * 2;         // 128x608 f16
static constexpr size_t SZ_QKVW  = (size_t)2 * NH * DH * DH * 2;   // per matrix, both blocks
static constexpr size_t SZ_W1    = (size_t)2 * FF * HID * 2;
static constexpr size_t SZ_W2    = (size_t)2 * HID * FF * 2;
static constexpr size_t SZ_HW    = (size_t)OUTD * HID * 2;

static constexpr size_t OFF_OUT  = 0;
static constexpr size_t OFF_XLN  = OFF_OUT + SZ_OUT;
static constexpr size_t OFF_Q    = OFF_XLN + SZ_XLN;
static constexpr size_t OFF_K    = OFF_Q + SZ_QKV;
static constexpr size_t OFF_VT   = OFF_K + SZ_QKV;
static constexpr size_t OFF_MLPH = OFF_VT + SZ_QKV;
static constexpr size_t OFF_LOG  = OFF_MLPH + SZ_MLPH;
static constexpr size_t OFF_POS  = OFF_LOG + SZ_LOG;
static constexpr size_t OFF_MAP16 = OFF_POS + SZ_POS;
static constexpr size_t OFF_QW16  = OFF_MAP16 + SZ_MAP16;
static constexpr size_t OFF_KW16  = OFF_QW16 + SZ_QKVW;
static constexpr size_t OFF_VW16  = OFF_KW16 + SZ_QKVW;
static constexpr size_t OFF_W116  = OFF_VW16 + SZ_QKVW;
static constexpr size_t OFF_W216  = OFF_W116 + SZ_W1;
static constexpr size_t OFF_HW16  = OFF_W216 + SZ_W2;

extern "C" void kernel_launch(void* const* d_in, const int* in_sizes, int n_in,
                              void* d_out, int out_size, void* d_ws, size_t ws_size,
                              hipStream_t stream) {
  const float* images = (const float*)d_in[0];
  const float* mapW   = (const float*)d_in[1];
  const float* mapb   = (const float*)d_in[2];
  const float* cls    = (const float*)d_in[3];
  const float* qW     = (const float*)d_in[4];
  const float* qb     = (const float*)d_in[5];
  const float* kW     = (const float*)d_in[6];
  const float* kb     = (const float*)d_in[7];
  const float* vW     = (const float*)d_in[8];
  const float* vb     = (const float*)d_in[9];
  const float* ln1g   = (const float*)d_in[10];
  const float* ln1b   = (const float*)d_in[11];
  const float* ln2g   = (const float*)d_in[12];
  const float* ln2b   = (const float*)d_in[13];
  const float* w1     = (const float*)d_in[14];
  const float* b1     = (const float*)d_in[15];
  const float* w2     = (const float*)d_in[16];
  const float* b2     = (const float*)d_in[17];
  const float* headW  = (const float*)d_in[18];
  const float* headb  = (const float*)d_in[19];

  char* ws = (char*)d_ws;
  float*     outbuf = (float*)(ws + OFF_OUT);
  _Float16*  xln    = (_Float16*)(ws + OFF_XLN);
  _Float16*  qbuf   = (_Float16*)(ws + OFF_Q);
  _Float16*  kbuf   = (_Float16*)(ws + OFF_K);
  _Float16*  vT     = (_Float16*)(ws + OFF_VT);
  _Float16*  mlph   = (_Float16*)(ws + OFF_MLPH);
  float*     logits = (float*)(ws + OFF_LOG);
  float*     posb   = (float*)(ws + OFF_POS);
  _Float16*  map16  = (_Float16*)(ws + OFF_MAP16);
  _Float16*  qW16   = (_Float16*)(ws + OFF_QW16);
  _Float16*  kW16   = (_Float16*)(ws + OFF_KW16);
  _Float16*  vW16   = (_Float16*)(ws + OFF_VW16);
  _Float16*  w116   = (_Float16*)(ws + OFF_W116);
  _Float16*  w216   = (_Float16*)(ws + OFF_W216);
  _Float16*  hW16   = (_Float16*)(ws + OFF_HW16);
  float*     prob   = (float*)d_out;

  // once-per-call prep: zero padded q/k/vT, POS table, f16 weight images
  k_zero<<<4096, 256, 0, stream>>>((uint4*)(ws + OFF_Q), (long)(3 * SZ_QKV / 16));
  k_pos<<<SEQL, 128, 0, stream>>>(posb);
  k_cvt_map<<<128, 256, 0, stream>>>(mapW, map16);
  k_cvt<<<64, 256, 0, stream>>>(qW, qW16, (long)2 * NH * DH * DH);
  k_cvt<<<64, 256, 0, stream>>>(kW, kW16, (long)2 * NH * DH * DH);
  k_cvt<<<64, 256, 0, stream>>>(vW, vW16, (long)2 * NH * DH * DH);
  k_cvt<<<256, 256, 0, stream>>>(w1, w116, (long)2 * FF * HID);
  k_cvt<<<256, 256, 0, stream>>>(w2, w216, (long)2 * HID * FF);
  k_cvt<<<256, 256, 0, stream>>>(headW, hW16, (long)OUTD * HID);

  k_patch<<<512, 256, 0, stream>>>(images, map16, mapb, posb, outbuf);
  k_cls<<<BATCH, 128, 0, stream>>>(cls, posb, outbuf);

  for (int blk = 0; blk < 2; ++blk) {
    long wq = (long)blk * NH * DH * DH;
    k_ln<<<8224, 256, 0, stream>>>(outbuf, ln1g + blk * HID, ln1b + blk * HID, xln);
    k_qkv<<<514, 256, 0, stream>>>(xln, qW16 + wq, kW16 + wq, vW16 + wq,
                                   qb + (long)blk * NH * DH, kb + (long)blk * NH * DH,
                                   vb + (long)blk * NH * DH, qbuf, kbuf, vT);
    dim3 ag(BATCH * NH, 9);
    k_attn<<<ag, 64, 0, stream>>>(qbuf, kbuf, vT, outbuf);
    k_ln<<<8224, 256, 0, stream>>>(outbuf, ln2g + blk * HID, ln2b + blk * HID, xln);
    k_mlp1<<<514, 256, 0, stream>>>(xln, w116 + (long)blk * FF * HID,
                                    b1 + (long)blk * FF, mlph);
    k_mlp2<<<514, 256, 0, stream>>>(mlph, w216 + (long)blk * HID * FF,
                                    b2 + (long)blk * HID, outbuf);
  }

  k_head<<<16, 256, 0, stream>>>(outbuf, hW16, headb, logits);
  k_softmax<<<BATCH, 256, 0, stream>>>(logits, prob);
}
